// MultiheadedAttention_68882685493836
// MI455X (gfx1250) — matrix operand order repaired and checked by reading, not compiled, not run on hardware
//
#include <hip/hip_runtime.h>
#include <hip/hip_bf16.h>

// ---------------------------------------------------------------------------
// MultiheadedAttention for MI455X (gfx1250, wave32, WMMA).
//   Q/K/V proj (bf16 WMMA GEMM) -> flash attention (bf16 WMMA, async-to-LDS
//   staging) -> out proj.
// Workspace: 4 * B*S*D bf16 = 64 MB.
// ---------------------------------------------------------------------------

#define B_    4
#define S_    2048
#define D_    1024
#define H_    16
#define DK_   64
#define MROWS 8192          // B_*S_

typedef __bf16 bf16;
typedef __attribute__((ext_vector_type(8)))  __bf16 bf16x8;
typedef __attribute__((ext_vector_type(16))) __bf16 bf16x16;
typedef __attribute__((ext_vector_type(8)))  float  f32x8;
typedef __attribute__((ext_vector_type(4)))  float  f32x4;

union Frag {
  bf16x16 v;
  bf16x8  h[2];
};

__device__ __forceinline__ f32x8 wmma_bf16(bf16x16 a, bf16x16 b, f32x8 c) {
  // D = A(16x32 bf16) * B(32x16 bf16) + C(16x16 f32)
  return __builtin_amdgcn_wmma_f32_16x16x32_bf16(
      /*neg_a=*/false, a, /*neg_b=*/false, b,
      /*c_mod=*/(short)0, c, /*reuse_a=*/false, /*reuse_b=*/false);
}

// ---- CDNA5 async global->LDS copy (ASYNCcnt-tracked, ISA 10/15.18) --------
// LDS dest address: generic shared pointer truncated to 32 bits (the LDS
// aperture maps addr[31:0] directly to the wave's LDS offset, ISA 10.2).
__device__ __forceinline__ void async_load_b128(const void* gsrc, void* lds) {
  unsigned laddr = (unsigned)(size_t)lds;
  asm volatile("global_load_async_to_lds_b128 %0, %1, off"
               :: "v"(laddr), "v"(gsrc) : "memory");
}
__device__ __forceinline__ void wait_asynccnt0() {
  asm volatile("s_wait_asynccnt 0x0" ::: "memory");
}

// ===========================================================================
// Kernel 1: Y_head[b][h][s][d] = (X @ W + bias)  (bf16 out, head-major)
//   Tile: 128(M) x 64(N), K-step 32. 8 waves, each wave owns a 32x32 subtile
//   (2x2 WMMA 16x16 tiles).
// ===========================================================================
#define TM 128
#define TN 64
#define TK 32
#define LDA_S 40   // 32 + 8 pad (bf16 elems)
#define LDB_S 40

__global__ __launch_bounds__(256) void qkv_proj_kernel(
    const float* __restrict__ X,    // [MROWS][D_] fp32
    const float* __restrict__ W,    // [D_][D_]   fp32
    const float* __restrict__ bias, // [D_]
    bf16* __restrict__ Out)         // [B_*H_][S_][DK_]
{
  __shared__ bf16 Ash[TM * LDA_S];
  __shared__ bf16 Bsh[TN * LDB_S];   // stored transposed: Bsh[n][k]

  const int n0   = blockIdx.x * TN;
  const int m0   = blockIdx.y * TM;
  const int t    = threadIdx.x;
  const int lane = t & 31;
  const int wave = t >> 5;
  const int wm   = wave & 3;   // M sub-block 0..3 (x32)
  const int wn   = wave >> 2;  // N sub-block 0..1 (x32)
  const int sub  = lane >> 4;  // lane half
  const int l16  = lane & 15;

  f32x8 acc[2][2] = {};

  const int ar = t >> 1;            // A-tile row this thread stages
  const int ac = (t & 1) * 16;      // A-tile col (16 floats)
  const int bk = t >> 3;            // B-tile row (k)
  const int bn = (t & 7) * 8;       // B-tile col group (8 floats)

  for (int k0 = 0; k0 < D_; k0 += TK) {
    __syncthreads();
    // ---- stage A tile (fp32 -> bf16) -----------------------------------
    {
      const float* src = X + (size_t)(m0 + ar) * D_ + k0 + ac;
      if (k0 + TK < D_) __builtin_prefetch(src + TK, 0, 1);
      f32x4 a0 = *(const f32x4*)(src + 0);
      f32x4 a1 = *(const f32x4*)(src + 4);
      f32x4 a2 = *(const f32x4*)(src + 8);
      f32x4 a3 = *(const f32x4*)(src + 12);
      bf16x8 lo, hi;
#pragma unroll
      for (int i = 0; i < 4; ++i) {
        lo[i]     = (bf16)a0[i];
        lo[i + 4] = (bf16)a1[i];
        hi[i]     = (bf16)a2[i];
        hi[i + 4] = (bf16)a3[i];
      }
      bf16* dst = &Ash[ar * LDA_S + ac];
      *(bf16x8*)(dst)     = lo;
      *(bf16x8*)(dst + 8) = hi;
    }
    // ---- stage B tile transposed (fp32 -> bf16) ------------------------
    {
      const float* src = W + (size_t)(k0 + bk) * D_ + n0 + bn;
      f32x4 b0 = *(const f32x4*)(src + 0);
      f32x4 b1 = *(const f32x4*)(src + 4);
#pragma unroll
      for (int i = 0; i < 4; ++i) {
        Bsh[(bn + i)     * LDB_S + bk] = (bf16)b0[i];
        Bsh[(bn + 4 + i) * LDB_S + bk] = (bf16)b1[i];
      }
    }
    __syncthreads();

    // ---- fragments + WMMA ----------------------------------------------
    Frag afr[2], bfr[2];
#pragma unroll
    for (int mt = 0; mt < 2; ++mt) {
      const bf16* ap = &Ash[(wm * 32 + mt * 16 + l16) * LDA_S + sub * 8];
      afr[mt].h[0] = *(const bf16x8*)(ap);
      afr[mt].h[1] = *(const bf16x8*)(ap + 16);
    }
#pragma unroll
    for (int nt = 0; nt < 2; ++nt) {
      const bf16* bp = &Bsh[(wn * 32 + nt * 16 + l16) * LDB_S + sub * 8];
      bfr[nt].h[0] = *(const bf16x8*)(bp);
      bfr[nt].h[1] = *(const bf16x8*)(bp + 16);
    }
#pragma unroll
    for (int mt = 0; mt < 2; ++mt)
#pragma unroll
      for (int nt = 0; nt < 2; ++nt)
        acc[mt][nt] = wmma_bf16(afr[mt].v, bfr[nt].v, acc[mt][nt]);
  }

  // ---- epilogue: scatter to head-major [b*H+h][s][d] -------------------
  const int h = n0 >> 6;   // TN==64 == dk, tile aligned to one head
#pragma unroll
  for (int mt = 0; mt < 2; ++mt) {
#pragma unroll
    for (int nt = 0; nt < 2; ++nt) {
      const int ng = n0 + wn * 32 + nt * 16 + l16;
      const float bval = bias[ng];
      const int d = ng & 63;
#pragma unroll
      for (int j = 0; j < 8; ++j) {
        const int mg = m0 + wm * 32 + mt * 16 + j + sub * 8;
        const int b  = mg >> 11;      // / S_
        const int s  = mg & 2047;
        Out[((size_t)(b * H_ + h) * S_ + s) * DK_ + d] =
            (bf16)(acc[mt][nt][j] + bval);
      }
    }
  }
}

// ===========================================================================
// Kernel 2: flash attention per (b,h, 128-query block).
//   8 waves * 16 query rows. Key blocks of 64. Online softmax.
//   Q block and K tiles staged with GLOBAL_LOAD_ASYNC_TO_LDS_B128.
// ===========================================================================
#define QB   128
#define KB   64
#define KV_S 72   // 64 + 8 pad (bf16 elems)

__global__ __launch_bounds__(256) void attn_kernel(
    const bf16* __restrict__ qp,    // [B_*H_][S_][DK_]
    const bf16* __restrict__ kp,
    const bf16* __restrict__ vp,
    const unsigned char* __restrict__ mask, // [B_][S_] bool
    bf16* __restrict__ ctx)         // [B_][S_][H_][DK_]
{
  __shared__ bf16 qsh[QB * KV_S];       // 18 KB
  __shared__ bf16 ksh[KB * KV_S];       //  9 KB  (row-major [key][d])
  __shared__ bf16 vtsh[DK_ * KV_S];     //  9 KB  (transposed [d][key])
  __shared__ bf16 psh[8 * 16 * KV_S];   // 18 KB  per-wave P strips

  const int bh = blockIdx.y;
  const int b  = bh >> 4;
  const int h  = bh & 15;
  const int q0 = blockIdx.x * QB;
  const int t    = threadIdx.x;
  const int lane = t & 31;
  const int wave = t >> 5;
  const int sub  = lane >> 4;
  const int l16  = lane & 15;

  const bf16* qbase = qp + ((size_t)bh * S_ + q0) * DK_;
  const bf16* kbase = kp + (size_t)bh * S_ * DK_;
  const bf16* vbase = vp + (size_t)bh * S_ * DK_;
  const unsigned char* mrow = mask + (size_t)b * S_;

  // ---- stage the 128x64 query block (async global->LDS) ----------------
  {
    const int r = t >> 1, c = (t & 1) * 32;
    const bf16* src = qbase + r * DK_ + c;
    bf16* dst = &qsh[r * KV_S + c];
    async_load_b128(src + 0,  dst + 0);
    async_load_b128(src + 8,  dst + 8);
    async_load_b128(src + 16, dst + 16);
    async_load_b128(src + 24, dst + 24);
  }
  wait_asynccnt0();
  __syncthreads();

  // ---- q fragments kept in registers (2 chunks over dk=64) -------------
  Frag qfr[2];
  {
    const int qr = wave * 16 + l16;
#pragma unroll
    for (int cc = 0; cc < 2; ++cc) {
      const bf16* ap = &qsh[qr * KV_S + cc * 32 + sub * 8];
      qfr[cc].h[0] = *(const bf16x8*)(ap);
      qfr[cc].h[1] = *(const bf16x8*)(ap + 16);
    }
  }

  float m_i[8], l_i[8];
  f32x8 oacc[4] = {};
#pragma unroll
  for (int j = 0; j < 8; ++j) { m_i[j] = -1e30f; l_i[j] = 0.f; }

  bf16* prow = &psh[wave * 16 * KV_S];

  for (int kb = 0; kb < S_; kb += KB) {
    __syncthreads();
    // ---- stage K block (async row-major) & V block (manual transpose) --
    {
      const int key = t >> 2, c = (t & 3) * 16;
      const bf16* ksrc = kbase + (size_t)(kb + key) * DK_ + c;
      bf16* kdst = &ksh[key * KV_S + c];
      async_load_b128(ksrc + 0, kdst + 0);
      async_load_b128(ksrc + 8, kdst + 8);
      const bf16* vsrc = vbase + (size_t)(kb + key) * DK_ + c;
      bf16x8 v0 = *(const bf16x8*)(vsrc);
      bf16x8 v1 = *(const bf16x8*)(vsrc + 8);
#pragma unroll
      for (int i = 0; i < 8; ++i) {
        vtsh[(c + i)     * KV_S + key] = v0[i];
        vtsh[(c + 8 + i) * KV_S + key] = v1[i];
      }
    }
    wait_asynccnt0();
    __syncthreads();

    // ---- scores: S = (q . k) * 1/sqrt(dk), 4 N-tiles x 2 k-chunks ------
    f32x8 sc[4];
#pragma unroll
    for (int nt = 0; nt < 4; ++nt) {
      f32x8 c0 = {};
#pragma unroll
      for (int cc = 0; cc < 2; ++cc) {
        Frag bfr;
        const bf16* bp = &ksh[(nt * 16 + l16) * KV_S + cc * 32 + sub * 8];
        bfr.h[0] = *(const bf16x8*)(bp);
        bfr.h[1] = *(const bf16x8*)(bp + 16);
        c0 = wmma_bf16(qfr[cc].v, bfr.v, c0);
      }
      const float madd = mrow[kb + nt * 16 + l16] ? 0.f : -1e9f;
#pragma unroll
      for (int j = 0; j < 8; ++j) sc[nt][j] = c0[j] * 0.125f + madd;
    }

    // ---- online softmax: per-row max over 64 keys ----------------------
    float alpha[8];
#pragma unroll
    for (int j = 0; j < 8; ++j) {
      float mx = fmaxf(fmaxf(sc[0][j], sc[1][j]), fmaxf(sc[2][j], sc[3][j]));
      mx = fmaxf(mx, __shfl_xor(mx, 1));
      mx = fmaxf(mx, __shfl_xor(mx, 2));
      mx = fmaxf(mx, __shfl_xor(mx, 4));
      mx = fmaxf(mx, __shfl_xor(mx, 8));
      const float mn = fmaxf(m_i[j], mx);
      alpha[j] = __expf(m_i[j] - mn);
      m_i[j] = mn;
    }
    // ---- p = exp(s - m); write P (bf16) into this wave's LDS strip -----
#pragma unroll
    for (int nt = 0; nt < 4; ++nt) {
#pragma unroll
      for (int j = 0; j < 8; ++j) {
        const float p = __expf(sc[nt][j] - m_i[j]);
        sc[nt][j] = p;
        prow[(j + sub * 8) * KV_S + nt * 16 + l16] = (bf16)p;
      }
    }
    // ---- row sums & l update -------------------------------------------
#pragma unroll
    for (int j = 0; j < 8; ++j) {
      float rs = sc[0][j] + sc[1][j] + sc[2][j] + sc[3][j];
      rs += __shfl_xor(rs, 1);
      rs += __shfl_xor(rs, 2);
      rs += __shfl_xor(rs, 4);
      rs += __shfl_xor(rs, 8);
      l_i[j] = l_i[j] * alpha[j] + rs;
    }
    // ---- rescale accumulator, then O += P @ V --------------------------
#pragma unroll
    for (int dt = 0; dt < 4; ++dt)
#pragma unroll
      for (int j = 0; j < 8; ++j) oacc[dt][j] = oacc[dt][j] * alpha[j];

#pragma unroll
    for (int cc = 0; cc < 2; ++cc) {   // key chunks of 32
      Frag pfr;
      const bf16* ap = &prow[l16 * KV_S + cc * 32 + sub * 8];
      pfr.h[0] = *(const bf16x8*)(ap);
      pfr.h[1] = *(const bf16x8*)(ap + 16);
#pragma unroll
      for (int dt = 0; dt < 4; ++dt) {
        Frag bfr;
        const bf16* bp = &vtsh[(dt * 16 + l16) * KV_S + cc * 32 + sub * 8];
        bfr.h[0] = *(const bf16x8*)(bp);
        bfr.h[1] = *(const bf16x8*)(bp + 16);
        oacc[dt] = wmma_bf16(pfr.v, bfr.v, oacc[dt]);
      }
    }
  }

  // ---- finalize: out = acc / l, store bf16 context [b][s][h*64+d] ------
#pragma unroll
  for (int dt = 0; dt < 4; ++dt) {
    const int d = dt * 16 + l16;
#pragma unroll
    for (int j = 0; j < 8; ++j) {
      const int s = q0 + wave * 16 + j + sub * 8;
      const float v = oacc[dt][j] / l_i[j];
      ctx[((size_t)(b * S_ + s) * H_ + h) * DK_ + d] = (bf16)v;
    }
  }
}

// ===========================================================================
// Kernel 3: Y = ctx(bf16) @ Wo + bo   (fp32 out, row-major [MROWS][D_])
// ===========================================================================
__global__ __launch_bounds__(256) void out_proj_kernel(
    const bf16* __restrict__ A,     // [MROWS][D_] bf16
    const float* __restrict__ W,    // [D_][D_]
    const float* __restrict__ bias, // [D_]
    float* __restrict__ Y)          // [MROWS][D_]
{
  __shared__ bf16 Ash[TM * LDA_S];
  __shared__ bf16 Bsh[TN * LDB_S];

  const int n0   = blockIdx.x * TN;
  const int m0   = blockIdx.y * TM;
  const int t    = threadIdx.x;
  const int lane = t & 31;
  const int wave = t >> 5;
  const int wm   = wave & 3;
  const int wn   = wave >> 2;
  const int sub  = lane >> 4;
  const int l16  = lane & 15;

  f32x8 acc[2][2] = {};

  const int ar = t >> 1;
  const int ac = (t & 1) * 16;
  const int bk = t >> 3;
  const int bn = (t & 7) * 8;

  for (int k0 = 0; k0 < D_; k0 += TK) {
    __syncthreads();
    {
      const bf16* src = A + (size_t)(m0 + ar) * D_ + k0 + ac;
      if (k0 + TK < D_) __builtin_prefetch(src + TK, 0, 1);
      bf16x8 lo = *(const bf16x8*)(src);
      bf16x8 hi = *(const bf16x8*)(src + 8);
      bf16* dst = &Ash[ar * LDA_S + ac];
      *(bf16x8*)(dst)     = lo;
      *(bf16x8*)(dst + 8) = hi;
    }
    {
      const float* src = W + (size_t)(k0 + bk) * D_ + n0 + bn;
      f32x4 b0 = *(const f32x4*)(src + 0);
      f32x4 b1 = *(const f32x4*)(src + 4);
#pragma unroll
      for (int i = 0; i < 4; ++i) {
        Bsh[(bn + i)     * LDB_S + bk] = (bf16)b0[i];
        Bsh[(bn + 4 + i) * LDB_S + bk] = (bf16)b1[i];
      }
    }
    __syncthreads();

    Frag afr[2], bfr[2];
#pragma unroll
    for (int mt = 0; mt < 2; ++mt) {
      const bf16* ap = &Ash[(wm * 32 + mt * 16 + l16) * LDA_S + sub * 8];
      afr[mt].h[0] = *(const bf16x8*)(ap);
      afr[mt].h[1] = *(const bf16x8*)(ap + 16);
    }
#pragma unroll
    for (int nt = 0; nt < 2; ++nt) {
      const bf16* bp = &Bsh[(wn * 32 + nt * 16 + l16) * LDB_S + sub * 8];
      bfr[nt].h[0] = *(const bf16x8*)(bp);
      bfr[nt].h[1] = *(const bf16x8*)(bp + 16);
    }
#pragma unroll
    for (int mt = 0; mt < 2; ++mt)
#pragma unroll
      for (int nt = 0; nt < 2; ++nt)
        acc[mt][nt] = wmma_bf16(afr[mt].v, bfr[nt].v, acc[mt][nt]);
  }

#pragma unroll
  for (int mt = 0; mt < 2; ++mt) {
#pragma unroll
    for (int nt = 0; nt < 2; ++nt) {
      const int ng = n0 + wn * 32 + nt * 16 + l16;
      const float bval = bias[ng];
#pragma unroll
      for (int j = 0; j < 8; ++j) {
        const int mg = m0 + wm * 32 + mt * 16 + j + sub * 8;
        Y[(size_t)mg * D_ + ng] = acc[mt][nt][j] + bval;
      }
    }
  }
}

// ===========================================================================
extern "C" void kernel_launch(void* const* d_in, const int* in_sizes, int n_in,
                              void* d_out, int out_size, void* d_ws, size_t ws_size,
                              hipStream_t stream) {
  const float* Q  = (const float*)d_in[0];
  const float* K  = (const float*)d_in[1];
  const float* V  = (const float*)d_in[2];
  const unsigned char* mask = (const unsigned char*)d_in[3]; // bool[B][1][S]
  const float* Wq = (const float*)d_in[4];
  const float* bq = (const float*)d_in[5];
  const float* Wk = (const float*)d_in[6];
  const float* bk = (const float*)d_in[7];
  const float* Wv = (const float*)d_in[8];
  const float* bv = (const float*)d_in[9];
  const float* Wo = (const float*)d_in[10];
  const float* bo = (const float*)d_in[11];
  float* Y = (float*)d_out;

  // Workspace layout (bf16): qp | kp | vp | ctx  -> 4 * 16 MB = 64 MB
  const size_t per = (size_t)B_ * H_ * S_ * DK_;
  bf16* qp  = (bf16*)d_ws;
  bf16* kp  = qp + per;
  bf16* vp  = kp + per;
  bf16* ctx = vp + per;

  dim3 blk(256);
  dim3 g1(D_ / TN, MROWS / TM);           // 16 x 64
  qkv_proj_kernel<<<g1, blk, 0, stream>>>(Q, Wq, bq, qp);
  qkv_proj_kernel<<<g1, blk, 0, stream>>>(K, Wk, bk, kp);
  qkv_proj_kernel<<<g1, blk, 0, stream>>>(V, Wv, bv, vp);

  dim3 g2(S_ / QB, B_ * H_);              // 16 x 64
  attn_kernel<<<g2, blk, 0, stream>>>(qp, kp, vp, mask, ctx);

  dim3 g3(D_ / TN, MROWS / TM);
  out_proj_kernel<<<g3, blk, 0, stream>>>(ctx, Wo, bo, Y);
}
